// DeformableTransformer_39453569581667
// MI455X (gfx1250) — compile-verified, matrix operand order checked by reading, not run on hardware
//
#include <hip/hip_runtime.h>
#include <math.h>

// ---------------------------------------------------------------------------
// Model constants (match reference)
// ---------------------------------------------------------------------------
#define Bm   4
#define NQm  100
#define Tm   25          // PT == TL == 25
#define Cm   256
#define NHm  8
#define DHm  32
#define NLm  4
#define NPm  4
#define DFFm 1024
#define LINm 21760
#define Ntok (Bm*NQm*Tm)     // 10000
#define BLIN (Bm*LINm)       // 87040

// ---------------------------------------------------------------------------
// WMMA / vector types
// ---------------------------------------------------------------------------
typedef __attribute__((ext_vector_type(16))) _Float16 v16h;
typedef __attribute__((ext_vector_type(4)))  _Float16 v4h;
typedef __attribute__((ext_vector_type(8)))  float    v8f;

#define GEMM_THREADS 128
#define LDS_STRIDE   40        // halves; 80B row stride keeps 8/16B alignment

// ---------------------------------------------------------------------------
// Generic GEMM:  C[N,M] = A[N,K] * W[M,K]^T + bias[M]   (optional ReLU)
// 64x64 tile / 128 threads / 4 waves; each wave: 32x32 via 2x2
// v_wmma_f32_16x16x32_f16.  fp32 operands staged to LDS as f16 with
// vectorized b128 global loads + b64 LDS stores, double-buffered so the
// next tile's global fetch overlaps the current tile's WMMAs.
// Requires: K % 32 == 0, M % 64 == 0 (true for all layers here).
// ---------------------------------------------------------------------------
__global__ void __launch_bounds__(GEMM_THREADS)
gemm_bias_wmma(const float* __restrict__ A, const float* __restrict__ W,
               const float* __restrict__ bias, float* __restrict__ C,
               int Nrows, int K, int M, int relu)
{
    __shared__ _Float16 As[2][64][LDS_STRIDE];
    __shared__ _Float16 Bs[2][64][LDS_STRIDE];

    const int tid  = threadIdx.x;
    const int lane = tid & 31;
    const int wave = tid >> 5;
    const int wy   = wave >> 1;
    const int wx   = wave & 1;
    const int rowBlock = blockIdx.x * 64;
    const int colBlock = blockIdx.y * 64;

    // ---- per-thread staging coordinates: 512 float4 per tile / 128 thr = 4
    int   lr[4], lc4[4];
    float mskA[4];
    const float4* Arow[4];
    const float4* Brow[4];
    #pragma unroll
    for (int i = 0; i < 4; ++i) {
        int idx = tid + i * GEMM_THREADS;    // 0..511
        int r   = idx >> 3;                  // 0..63
        int c4  = idx & 7;                   // 0..7  (float4 column)
        lr[i] = r; lc4[i] = c4;
        int gr  = rowBlock + r;
        mskA[i] = (gr < Nrows) ? 1.f : 0.f;
        if (gr >= Nrows) gr = Nrows - 1;     // clamped, always-legal load
        Arow[i] = (const float4*)(A + (size_t)gr * K);
        Brow[i] = (const float4*)(W + (size_t)(colBlock + r) * K);  // M%64==0
    }

    v8f acc00 = {}, acc01 = {}, acc10 = {}, acc11 = {};

    // 16-bit A-matrix 16x32 fragment layout (ISA 7.12.2):
    // lanes 0-15: M=lane, K {0..7,16..23}; lanes 16-31: M=lane-16, K {8..15,24..31}
    const int l16 = lane & 15;
    const int kh  = (lane & 16) ? 8 : 0;

    float4 ra[4], rb[4];

    // ---- prologue: fetch + stage tile 0
    #pragma unroll
    for (int i = 0; i < 4; ++i) {
        ra[i] = Arow[i][lc4[i]];
        rb[i] = Brow[i][lc4[i]];
    }
    #pragma unroll
    for (int i = 0; i < 4; ++i) {
        v4h ha, hb;
        ha[0] = (_Float16)(ra[i].x * mskA[i]); ha[1] = (_Float16)(ra[i].y * mskA[i]);
        ha[2] = (_Float16)(ra[i].z * mskA[i]); ha[3] = (_Float16)(ra[i].w * mskA[i]);
        hb[0] = (_Float16)rb[i].x; hb[1] = (_Float16)rb[i].y;
        hb[2] = (_Float16)rb[i].z; hb[3] = (_Float16)rb[i].w;
        *(v4h*)&As[0][lr[i]][lc4[i] * 4] = ha;
        *(v4h*)&Bs[0][lr[i]][lc4[i] * 4] = hb;
    }
    __syncthreads();

    int buf = 0;
    for (int k0 = 0; k0 < K; k0 += 32) {
        const bool more = (k0 + 32) < K;
        // ---- fetch next tile from global (overlaps with WMMAs below)
        if (more) {
            int f4 = (k0 + 32) >> 2;
            #pragma unroll
            for (int i = 0; i < 4; ++i) {
                ra[i] = Arow[i][f4 + lc4[i]];
                rb[i] = Brow[i][f4 + lc4[i]];
            }
        }
        // ---- build fragments and issue 4 WMMAs from LDS[buf]
        v16h a0, a1, b0, b1;
        #pragma unroll
        for (int i = 0; i < 8; ++i) {
            a0[i]     = As[buf][wy * 32 +      l16][kh + i];
            a0[8 + i] = As[buf][wy * 32 +      l16][16 + kh + i];
            a1[i]     = As[buf][wy * 32 + 16 + l16][kh + i];
            a1[8 + i] = As[buf][wy * 32 + 16 + l16][16 + kh + i];
            b0[i]     = Bs[buf][wx * 32 +      l16][kh + i];
            b0[8 + i] = Bs[buf][wx * 32 +      l16][16 + kh + i];
            b1[i]     = Bs[buf][wx * 32 + 16 + l16][kh + i];
            b1[8 + i] = Bs[buf][wx * 32 + 16 + l16][16 + kh + i];
        }
        acc00 = __builtin_amdgcn_wmma_f32_16x16x32_f16(false, a0, false, b0, (short)0, acc00, false, false);
        acc01 = __builtin_amdgcn_wmma_f32_16x16x32_f16(false, a0, false, b1, (short)0, acc01, false, false);
        acc10 = __builtin_amdgcn_wmma_f32_16x16x32_f16(false, a1, false, b0, (short)0, acc10, false, false);
        acc11 = __builtin_amdgcn_wmma_f32_16x16x32_f16(false, a1, false, b1, (short)0, acc11, false, false);
        // ---- stage next tile into the other buffer
        if (more) {
            #pragma unroll
            for (int i = 0; i < 4; ++i) {
                v4h ha, hb;
                ha[0] = (_Float16)(ra[i].x * mskA[i]); ha[1] = (_Float16)(ra[i].y * mskA[i]);
                ha[2] = (_Float16)(ra[i].z * mskA[i]); ha[3] = (_Float16)(ra[i].w * mskA[i]);
                hb[0] = (_Float16)rb[i].x; hb[1] = (_Float16)rb[i].y;
                hb[2] = (_Float16)rb[i].z; hb[3] = (_Float16)rb[i].w;
                *(v4h*)&As[buf ^ 1][lr[i]][lc4[i] * 4] = ha;
                *(v4h*)&Bs[buf ^ 1][lr[i]][lc4[i] * 4] = hb;
            }
        }
        __syncthreads();
        buf ^= 1;
    }

    // C/D 16x16 f32 layout: VGPR r -> M = r + 8*(lane>=16), N = lane&15.
    const int mOff = (lane & 16) ? 8 : 0;
    const int nIdx = lane & 15;
    #pragma unroll
    for (int r = 0; r < 8; ++r) {
        int gr0 = rowBlock + wy * 32 + r + mOff;
        int gr1 = gr0 + 16;
        int gc0 = colBlock + wx * 32 + nIdx;
        int gc1 = gc0 + 16;
        float o;
        if (gr0 < Nrows) {
            o = acc00[r] + bias[gc0]; if (relu) o = fmaxf(o, 0.f); C[(size_t)gr0 * M + gc0] = o;
            o = acc01[r] + bias[gc1]; if (relu) o = fmaxf(o, 0.f); C[(size_t)gr0 * M + gc1] = o;
        }
        if (gr1 < Nrows) {
            o = acc10[r] + bias[gc0]; if (relu) o = fmaxf(o, 0.f); C[(size_t)gr1 * M + gc0] = o;
            o = acc11[r] + bias[gc1]; if (relu) o = fmaxf(o, 0.f); C[(size_t)gr1 * M + gc1] = o;
        }
    }
}

// ---------------------------------------------------------------------------
// Elementwise add:  o = a + b
// ---------------------------------------------------------------------------
__global__ void add_kernel(const float* __restrict__ a, const float* __restrict__ b,
                           float* __restrict__ o, int n)
{
    int i = blockIdx.x * blockDim.x + threadIdx.x;
    if (i < n) o[i] = a[i] + b[i];
}

// ---------------------------------------------------------------------------
// Fused residual + LayerNorm over C=256. One wave32 per token, 8 elems/lane.
// ---------------------------------------------------------------------------
__global__ void __launch_bounds__(32)
add_ln_kernel(const float* __restrict__ x, const float* __restrict__ y,
              const float* __restrict__ gamma, const float* __restrict__ beta,
              float* __restrict__ out)
{
    const int n    = blockIdx.x;
    const int lane = threadIdx.x;
    float v[8];
    float s = 0.f, s2 = 0.f;
    #pragma unroll
    for (int i = 0; i < 8; ++i) {
        int c = lane + i * 32;
        float f = x[(size_t)n * Cm + c] + y[(size_t)n * Cm + c];
        v[i] = f; s += f; s2 += f * f;
    }
    #pragma unroll
    for (int o = 16; o; o >>= 1) {
        s  += __shfl_xor(s,  o, 32);
        s2 += __shfl_xor(s2, o, 32);
    }
    float mean = s * (1.f / Cm);
    float var  = s2 * (1.f / Cm) - mean * mean;
    float inv  = rsqrtf(var + 1e-5f);
    #pragma unroll
    for (int i = 0; i < 8; ++i) {
        int c = lane + i * 32;
        out[(size_t)n * Cm + c] = (v[i] - mean) * inv * gamma[c] + beta[c];
    }
}

// ---------------------------------------------------------------------------
// Flash-style MHA. qk[N,512] holds Q (cols 0..255) and K (cols 256..511)
// head-packed; v[N,256]. Row mapping for member m of group g:
//   row = (g/div)*aMul + (g%div)*cMul + m*mStride
// One wave per (group, head, query-row); lanes = 32 head dims; online softmax.
// ---------------------------------------------------------------------------
__global__ void __launch_bounds__(128)
attn_flash_kernel(const float* __restrict__ qk, const float* __restrict__ v,
                  float* __restrict__ out, int L, int div, int aMul, int cMul,
                  int mStride, float scale)
{
    const int g = blockIdx.x;
    const int h = blockIdx.y;
    const int l = blockIdx.z * blockDim.y + threadIdx.y;
    if (l >= L) return;
    const int lane = threadIdx.x;

    const int gbase = (g / div) * aMul + (g % div) * cMul;
    const int lrow  = gbase + l * mStride;

    float qd = qk[(size_t)lrow * 512 + h * 32 + lane];

    float rmax = -1e30f, rsum = 0.f, acc = 0.f;
    for (int m = 0; m < L; ++m) {
        int mrow = gbase + m * mStride;
        float p = qd * qk[(size_t)mrow * 512 + 256 + h * 32 + lane];
        #pragma unroll
        for (int o = 16; o; o >>= 1) p += __shfl_xor(p, o, 32);
        float s    = p * scale;
        float nmax = fmaxf(rmax, s);
        float corr = __expf(rmax - nmax);
        float e    = __expf(s - nmax);
        rsum = rsum * corr + e;
        acc  = acc  * corr + e * v[(size_t)mrow * Cm + h * 32 + lane];
        rmax = nmax;
    }
    out[(size_t)lrow * Cm + h * 32 + lane] = acc / rsum;
}

// ---------------------------------------------------------------------------
// MS deformable attention sampling.
// val[BLIN,256] (head-packed), off[N,256], awl[N,128], ref[B,NQ,NL,2].
// Block (32,8): one wave per head; lane = head dim. Per-lane 16-way softmax
// of attention logits + 4-level bilinear gather with zero padding.
// ---------------------------------------------------------------------------
__global__ void __launch_bounds__(256)
msda_sample_kernel(const float* __restrict__ val, const float* __restrict__ off,
                   const float* __restrict__ awl, const float* __restrict__ ref,
                   float* __restrict__ out)
{
    const int n    = blockIdx.x;
    const int h    = threadIdx.y;
    const int lane = threadIdx.x;

    const int b = n / (NQm * Tm);
    const int q = (n / Tm) % NQm;

    float lg[16];
    float mx = -1e30f;
    #pragma unroll
    for (int i = 0; i < 16; ++i) {
        lg[i] = awl[(size_t)n * 128 + h * 16 + i];
        mx = fmaxf(mx, lg[i]);
    }
    float se = 0.f;
    #pragma unroll
    for (int i = 0; i < 16; ++i) { lg[i] = __expf(lg[i] - mx); se += lg[i]; }
    const float invse = 1.f / se;

    const int Hs[4]    = {128, 64, 32, 16};
    const int Wsz[4]   = {128, 64, 32, 16};
    const int start[4] = {0, 16384, 20480, 21504};

    float acc = 0.f;
    #pragma unroll
    for (int l = 0; l < NLm; ++l) {
        const int   Hl = Hs[l], Wl = Wsz[l];
        const float rx = ref[(((size_t)(b * NQm + q)) * NLm + l) * 2 + 0];
        const float ry = ref[(((size_t)(b * NQm + q)) * NLm + l) * 2 + 1];
        const size_t vbase = ((size_t)(b * LINm + start[l])) * Cm + h * 32 + lane;
        #pragma unroll
        for (int p = 0; p < NPm; ++p) {
            const int oidx = (((h * NLm + l) * NPm) + p) * 2;
            float ox = off[(size_t)n * 256 + oidx + 0];
            float oy = off[(size_t)n * 256 + oidx + 1];
            float lx = rx + ox / (float)Wl;
            float ly = ry + oy / (float)Hl;
            float xx = lx * Wl - 0.5f;
            float yy = ly * Hl - 0.5f;
            float x0f = floorf(xx), y0f = floorf(yy);
            float fx = xx - x0f, fy = yy - y0f;
            int x0i = (int)x0f, y0i = (int)y0f;
            float smp = 0.f;
            #pragma unroll
            for (int c = 0; c < 4; ++c) {
                int dx = c & 1, dy = c >> 1;
                float wgt = (dx ? fx : 1.f - fx) * (dy ? fy : 1.f - fy);
                int xi = x0i + dx, yi = y0i + dy;
                bool valid = (xi >= 0) && (xi <= Wl - 1) && (yi >= 0) && (yi <= Hl - 1);
                int xc = xi < 0 ? 0 : (xi > Wl - 1 ? Wl - 1 : xi);
                int yc = yi < 0 ? 0 : (yi > Hl - 1 ? Hl - 1 : yi);
                float vv = val[vbase + (size_t)(yc * Wl + xc) * Cm];
                smp += valid ? wgt * vv : 0.f;
            }
            acc += lg[l * NPm + p] * invse * smp;
        }
    }
    out[(size_t)n * Cm + h * 32 + lane] = acc;
}

// ---------------------------------------------------------------------------
// Host launcher
// ---------------------------------------------------------------------------
extern "C" void kernel_launch(void* const* d_in, const int* in_sizes, int n_in,
                              void* d_out, int out_size, void* d_ws, size_t ws_size,
                              hipStream_t stream)
{
    (void)in_sizes; (void)n_in; (void)out_size; (void)ws_size;

    const float* tgt    = (const float*)d_in[0];
    const float* qpos   = (const float*)d_in[1];
    const float* tgt_t  = (const float*)d_in[2];
    const float* qpos_t = (const float*)d_in[3];
    const float* ref    = (const float*)d_in[4];
    const float* src    = (const float*)d_in[5];
    // d_in[6..8]: spatial shapes / level starts (static), pad mask (all false)

    float* ws = (float*)d_ws;
    float* W0 = ws;                          // N*256  (q / residual stream)
    float* W1 = W0 + (size_t)Ntok * Cm;      // N*512  (QK pack / offsets)
    float* W2 = W1 + (size_t)Ntok * 512;     // N*256  (V / gemm outs / aw logits)
    float* W3 = W2 + (size_t)Ntok * Cm;      // N*256  (attention / sample out)
    float* W4 = W3 + (size_t)Ntok * Cm;      // N*256  (residual stream)
    float* W6 = W4 + (size_t)Ntok * Cm;      // max(BLIN*256, N*1024): value / FFN hidden

    const float scale = 1.f / sqrtf((float)DHm);

    auto gemm = [&](const float* A, const float* Wt, const float* bias, float* C,
                    int Nrows, int K, int M, int relu) {
        dim3 grid((Nrows + 63) / 64, M / 64);
        gemm_bias_wmma<<<grid, GEMM_THREADS, 0, stream>>>(A, Wt, bias, C, Nrows, K, M, relu);
    };
    auto addv = [&](const float* a, const float* b, float* o, int n) {
        add_kernel<<<(n + 255) / 256, 256, 0, stream>>>(a, b, o, n);
    };
    auto add_ln = [&](const float* x, const float* y, const float* g, const float* bt, float* o) {
        add_ln_kernel<<<Ntok, 32, 0, stream>>>(x, y, g, bt, o);
    };

    for (int br = 0; br < 2; ++br) {
        const int pb = 9 + br * 28;
        const float* x0  = br ? tgt_t  : tgt;
        const float* pos = br ? qpos_t : qpos;
        const float* in_iw = (const float*)d_in[pb + 0];
        const float* in_ib = (const float*)d_in[pb + 1];
        const float* in_ow = (const float*)d_in[pb + 2];
        const float* in_ob = (const float*)d_in[pb + 3];
        const float* ie_iw = (const float*)d_in[pb + 4];
        const float* ie_ib = (const float*)d_in[pb + 5];
        const float* ie_ow = (const float*)d_in[pb + 6];
        const float* ie_ob = (const float*)d_in[pb + 7];
        const float* vw  = (const float*)d_in[pb + 8];
        const float* vb  = (const float*)d_in[pb + 9];
        const float* sw  = (const float*)d_in[pb + 10];
        const float* sb  = (const float*)d_in[pb + 11];
        const float* aww = (const float*)d_in[pb + 12];
        const float* awb = (const float*)d_in[pb + 13];
        const float* pw  = (const float*)d_in[pb + 14];
        const float* pbv = (const float*)d_in[pb + 15];
        const float* w1  = (const float*)d_in[pb + 16];
        const float* b1  = (const float*)d_in[pb + 17];
        const float* w2  = (const float*)d_in[pb + 18];
        const float* b2  = (const float*)d_in[pb + 19];
        const float* ni_g = (const float*)d_in[pb + 20];
        const float* ni_b = (const float*)d_in[pb + 21];
        const float* ne_g = (const float*)d_in[pb + 22];
        const float* ne_b = (const float*)d_in[pb + 23];
        const float* nc_g = (const float*)d_in[pb + 24];
        const float* nc_b = (const float*)d_in[pb + 25];
        const float* n3_g = (const float*)d_in[pb + 26];
        const float* n3_b = (const float*)d_in[pb + 27];
        float* outp = (float*)d_out + (size_t)br * Ntok * Cm;

        // ---- intra self-attn (across point/char axis) ----
        addv(x0, pos, W0, Ntok * Cm);                          // q = tgt + qpos
        gemm(W0, in_iw,             in_ib,       W1, Ntok, Cm, 512, 0);  // Q|K
        gemm(x0, in_iw + 512 * Cm,  in_ib + 512, W2, Ntok, Cm, Cm,  0);  // V (from tgt)
        attn_flash_kernel<<<dim3(Bm * NQm, NHm, (Tm + 3) / 4), dim3(32, 4), 0, stream>>>(
            W1, W2, W3, Tm, Bm * NQm, 0, Tm, 1, scale);
        gemm(W3, in_ow, in_ob, W2, Ntok, Cm, Cm, 0);           // out proj
        add_ln(x0, W2, ni_g, ni_b, W4);                        // t

        // ---- inter self-attn (across proposals, transposed view) ----
        gemm(W4, ie_iw,            ie_ib,       W1, Ntok, Cm, 512, 0);
        gemm(W4, ie_iw + 512 * Cm, ie_ib + 512, W2, Ntok, Cm, Cm,  0);
        attn_flash_kernel<<<dim3(Bm * Tm, NHm, (NQm + 3) / 4), dim3(32, 4), 0, stream>>>(
            W1, W2, W3, NQm, Tm, NQm * Tm, 1, Tm, scale);
        gemm(W3, ie_ow, ie_ob, W2, Ntok, Cm, Cm, 0);
        add_ln(W4, W2, ne_g, ne_b, W0);                        // ti

        // ---- MS deformable cross attention ----
        addv(W0, pos, W4, Ntok * Cm);                          // query = ti + qpos
        gemm(W4, sw,  sb,  W1, Ntok, Cm, Cm,  0);              // sampling offsets
        gemm(W4, aww, awb, W2, Ntok, Cm, 128, 0);              // attention logits
        gemm(src, vw, vb,  W6, BLIN, Cm, Cm,  0);              // value projection
        msda_sample_kernel<<<Ntok, dim3(32, NHm), 0, stream>>>(W6, W1, W2, ref, W3);
        gemm(W3, pw, pbv, W2, Ntok, Cm, Cm, 0);                // output proj
        add_ln(W0, W2, nc_g, nc_b, W4);                        // t

        // ---- FFN ----
        gemm(W4, w1, b1, W6, Ntok, Cm,  DFFm, 1);              // relu(h)
        gemm(W6, w2, b2, W2, Ntok, DFFm, Cm,  0);
        add_ln(W4, W2, n3_g, n3_b, outp);                      // final LN -> output
    }
}
